// TelephoneAttentionBlock_2259152798576
// MI455X (gfx1250) — matrix operand order repaired and checked by reading, not compiled, not run on hardware
//
#include <hip/hip_runtime.h>

// ---------------------------------------------------------------------------
// Telephone (sliding-window) attention block for MI455X (gfx1250, wave32).
// bf16 WMMA (v_wmma_f32_16x16x32_bf16) everywhere, f32 accumulation/softmax.
// QKV GEMM stages B tiles via async global->LDS (ASYNCcnt); out-proj GEMM
// stages B tiles via the Tensor Data Mover (TENSORcnt); both double-buffered.
// ---------------------------------------------------------------------------

#define BB 4
#define SS 4096
#define DD 1024
#define HH 16
#define DH 64
#define WW 128
#define NB 32            // SS / WW
#define D3 3072          // 3*DD

typedef __attribute__((ext_vector_type(16))) __bf16       v16bf;
typedef __attribute__((ext_vector_type(8)))  float        v8f;
typedef __attribute__((ext_vector_type(4))) unsigned int  u32x4;
typedef __attribute__((ext_vector_type(2))) unsigned int  u32x2;
typedef __attribute__((ext_vector_type(4)))  float        f32x4;
typedef __attribute__((ext_vector_type(8)))  int          i32x8;
typedef __attribute__((ext_vector_type(4)))  int          i32x4;

union Frag { u32x4 u[2]; v16bf v; };

__device__ inline unsigned short f2bf(float f) {            // f32 -> bf16 RNE
    unsigned int u = __float_as_uint(f);
    u = (u + 0x7FFFu + ((u >> 16) & 1u)) >> 16;
    return (unsigned short)u;
}

__device__ inline v16bf load_frag(const unsigned short* p0, const unsigned short* p1) {
    Frag f;
    f.u[0] = *(const u32x4*)p0;
    f.u[1] = *(const u32x4*)p1;
    return f.v;
}

__device__ inline v8f wmma_bf16(v16bf a, v16bf b, v8f c) {
    return __builtin_amdgcn_wmma_f32_16x16x32_bf16(false, a, false, b, (short)0, c,
                                                   false, false);
}

__device__ inline unsigned int lds_off(const void* p) {     // generic->LDS offset
    return (unsigned int)(unsigned long long)p;
}

// Async global->LDS 16B copy (per-lane addresses), tracked by ASYNCcnt.
__device__ inline void async_copy_b128(unsigned int dst_lds, const void* src) {
    asm volatile("global_load_async_to_lds_b128 %0, %1, off"
                 :: "v"(dst_lds), "v"((unsigned long long)src)
                 : "memory");
}

// ---------------------------------------------------------------------------
// Kernel 1: RMSNorm (f32) -> bf16 activations. One block per row of 1024.
// ---------------------------------------------------------------------------
__global__ void __launch_bounds__(256)
rmsnorm_bf16_kernel(const float* __restrict__ x, const float* __restrict__ g,
                    unsigned short* __restrict__ xn) {
    __shared__ float red[256];
    const int row = blockIdx.x;
    const float* xr = x + (size_t)row * DD;
    const int c0 = threadIdx.x * 4;
    f32x4 xv = *(const f32x4*)(xr + c0);
    red[threadIdx.x] = xv.x * xv.x + xv.y * xv.y + xv.z * xv.z + xv.w * xv.w;
    __syncthreads();
    for (int s = 128; s > 0; s >>= 1) {
        if (threadIdx.x < s) red[threadIdx.x] += red[threadIdx.x + s];
        __syncthreads();
    }
    const float rn = rsqrtf(red[0] * (1.0f / DD) + 1e-6f);
    f32x4 gv = *(const f32x4*)(g + c0);
    unsigned short* out = xn + (size_t)row * DD + c0;
    out[0] = f2bf(xv.x * rn * gv.x);
    out[1] = f2bf(xv.y * rn * gv.y);
    out[2] = f2bf(xv.z * rn * gv.z);
    out[3] = f2bf(xv.w * rn * gv.w);
}

// ---------------------------------------------------------------------------
// Kernel 2: transpose f32 [K,N] weight -> bf16 [N,K]
// ---------------------------------------------------------------------------
__global__ void __launch_bounds__(256)
transpose_bf16_kernel(const float* __restrict__ in, unsigned short* __restrict__ out,
                      int K, int N) {
    const unsigned int idx = blockIdx.x * 256u + threadIdx.x;
    const unsigned int n = idx / (unsigned)K;
    const unsigned int k = idx % (unsigned)K;
    if (n < (unsigned)N) out[idx] = f2bf(in[(size_t)k * N + n]);
}

// ---------------------------------------------------------------------------
// Kernel 3: QKV GEMM with async-LDS B staging (double buffered).
// A: xn bf16 [16384,1024]. Bw: WqkvT bf16 [3072,1024] (N,K). C: [16384,3072].
// Block = 128 thr / 4 waves; tile 64(M)x64(N); B tile 64Nx32K = 4KB in LDS.
// ---------------------------------------------------------------------------
__global__ void __launch_bounds__(128)
qkv_gemm_kernel(const unsigned short* __restrict__ A,
                const unsigned short* __restrict__ Bw,
                unsigned short* __restrict__ C) {
    const int K = DD, N = D3, NIT = DD / 32;
    __shared__ unsigned short Bt[2][64 * 32];

    const int w = threadIdx.x >> 5, lane = threadIdx.x & 31;
    const int ln = lane & 15, hi = lane >> 4;
    const int rowA = blockIdx.y * 64 + w * 16 + ln;
    const int colBase = blockIdx.x * 64;
    const unsigned short* arow = A + (size_t)rowA * K;

    // staging map: thread t copies 32B of row n = t/2, half = t%2
    const int sn = threadIdx.x >> 1, sh = threadIdx.x & 1;
    const unsigned short* srow = Bw + (size_t)(colBase + sn) * K + sh * 16;

    v8f acc[4];
    for (int t = 0; t < 4; ++t) acc[t] = (v8f){};

    // prefetch tile 0
    {
        unsigned int d = lds_off(&Bt[0][sn * 32 + sh * 16]);
        async_copy_b128(d, srow);
        async_copy_b128(d + 16, srow + 8);
    }
#pragma unroll 1
    for (int i = 0; i < NIT; ++i) {
        const int cur = i & 1;
        if (i + 1 < NIT) {
            const unsigned short* s2 = srow + (i + 1) * 32;
            unsigned int d = lds_off(&Bt[cur ^ 1][sn * 32 + sh * 16]);
            async_copy_b128(d, s2);
            async_copy_b128(d + 16, s2 + 8);
            asm volatile("s_wait_asynccnt 0x2" ::: "memory");
        } else {
            asm volatile("s_wait_asynccnt 0x0" ::: "memory");
        }
        __syncthreads();
        const int ks = i * 32;
        v16bf af = load_frag(arow + ks + 16 * hi, arow + ks + 8 + 16 * hi);
#pragma unroll
        for (int t = 0; t < 4; ++t) {
            const unsigned short* bp = &Bt[cur][(t * 16 + ln) * 32 + 16 * hi];
            v16bf bf = load_frag(bp, bp + 8);
            acc[t] = wmma_bf16(af, bf, acc[t]);
        }
        __syncthreads();
    }
    const int crow0 = blockIdx.y * 64 + w * 16;
#pragma unroll
    for (int t = 0; t < 4; ++t)
#pragma unroll
        for (int r = 0; r < 8; ++r) {
            const int m = crow0 + r + 8 * hi;
            const int n = colBase + t * 16 + ln;
            C[(size_t)m * N + n] = f2bf(acc[t][r]);
        }
}

// ---------------------------------------------------------------------------
// Kernel 4: sliding-window attention.
// ---------------------------------------------------------------------------
__global__ void __launch_bounds__(128)
attn_kernel(const unsigned short* __restrict__ qkv, unsigned short* __restrict__ O) {
    __shared__ unsigned short Vt[DH * 2 * WW];      // [dh][256 keys], 32 KB
    __shared__ unsigned short P[4 * 16 * 2 * WW];   // per-wave [16][256], 32 KB

    const int idx = blockIdx.x;
    const int half = idx & 1;
    const int rest = idx >> 1;
    const int blk = rest & (NB - 1);
    const int h = (rest >> 5) & (HH - 1);
    const int b = rest >> 9;

    {   // stage V_ext transposed into LDS
        const int c = threadIdx.x & 15, jr = threadIdx.x >> 4;
        const int dh0 = c * 4;
        for (int it = 0; it < 32; ++it) {
            const int j = jr + it * 8;
            int grow = blk * WW + j - WW;
            if (grow < 0) grow = 0;
            const unsigned short* src =
                qkv + ((size_t)b * SS + grow) * D3 + 2 * DD + h * DH + dh0;
            u32x2 u = *(const u32x2*)src;
            Vt[(dh0 + 0) * 256 + j] = (unsigned short)(u.x & 0xffffu);
            Vt[(dh0 + 1) * 256 + j] = (unsigned short)(u.x >> 16);
            Vt[(dh0 + 2) * 256 + j] = (unsigned short)(u.y & 0xffffu);
            Vt[(dh0 + 3) * 256 + j] = (unsigned short)(u.y >> 16);
        }
    }
    __syncthreads();

    const int w = threadIdx.x >> 5, lane = threadIdx.x & 31;
    const int ln = lane & 15, hi = lane >> 4;
    const int qloc = half * 64 + w * 16;
    const int qrow = blk * WW + qloc + ln;
    const unsigned short* qp = qkv + ((size_t)b * SS + qrow) * D3 + h * DH;
    const v16bf qa0 = load_frag(qp + 16 * hi, qp + 8 + 16 * hi);
    const v16bf qa1 = load_frag(qp + 32 + 16 * hi, qp + 40 + 16 * hi);

    v8f s[16];
#pragma unroll
    for (int t = 0; t < 16; ++t) {
        int grow = blk * WW + t * 16 + ln - WW;
        if (grow < 0) grow = 0;
        const unsigned short* kp = qkv + ((size_t)b * SS + grow) * D3 + DD + h * DH;
        v16bf kb0 = load_frag(kp + 16 * hi, kp + 8 + 16 * hi);
        v16bf kb1 = load_frag(kp + 32 + 16 * hi, kp + 40 + 16 * hi);
        v8f z = (v8f){};
        z = wmma_bf16(qa0, kb0, z);
        s[t] = wmma_bf16(qa1, kb1, z);
    }

    const float scale = 0.125f;  // 1/sqrt(64)
    float rmax[8], rsum[8];
#pragma unroll
    for (int r = 0; r < 8; ++r) rmax[r] = -1e30f;
#pragma unroll
    for (int t = 0; t < 16; ++t) {
        const int j = t * 16 + ln;
#pragma unroll
        for (int r = 0; r < 8; ++r) {
            const int i = qloc + r + 8 * hi;
            const bool ok = (j > i) && (j <= i + WW) && (blk > 0 || j >= WW);
            const float v = ok ? s[t][r] * scale : -1e30f;
            s[t][r] = v;
            rmax[r] = fmaxf(rmax[r], v);
        }
    }
    for (int off = 1; off < 16; off <<= 1)
#pragma unroll
        for (int r = 0; r < 8; ++r)
            rmax[r] = fmaxf(rmax[r], __shfl_xor(rmax[r], off, 16));
#pragma unroll
    for (int r = 0; r < 8; ++r) rsum[r] = 0.0f;
#pragma unroll
    for (int t = 0; t < 16; ++t)
#pragma unroll
        for (int r = 0; r < 8; ++r) {
            const float p = __expf(s[t][r] - rmax[r]);
            s[t][r] = p;
            rsum[r] += p;
        }
    for (int off = 1; off < 16; off <<= 1)
#pragma unroll
        for (int r = 0; r < 8; ++r) rsum[r] += __shfl_xor(rsum[r], off, 16);

    unsigned short* Pw = P + w * 16 * 256;
#pragma unroll
    for (int t = 0; t < 16; ++t)
#pragma unroll
        for (int r = 0; r < 8; ++r)
            Pw[(r + 8 * hi) * 256 + t * 16 + ln] = f2bf(s[t][r]);
    asm volatile("s_wait_dscnt 0" ::: "memory");  // same-wave DS RAW fence

    v8f o[4];
#pragma unroll
    for (int t = 0; t < 4; ++t) o[t] = (v8f){};
    for (int ks = 0; ks < 256; ks += 32) {
        const unsigned short* pp = Pw + ln * 256 + ks;
        v16bf pa = load_frag(pp + 16 * hi, pp + 8 + 16 * hi);
#pragma unroll
        for (int ct = 0; ct < 4; ++ct) {
            const unsigned short* vp = Vt + (ct * 16 + ln) * 256 + ks + 16 * hi;
            v16bf vb = load_frag(vp, vp + 8);
            o[ct] = wmma_bf16(pa, vb, o[ct]);
        }
    }
#pragma unroll
    for (int ct = 0; ct < 4; ++ct)
#pragma unroll
        for (int r = 0; r < 8; ++r) {
            const float val = o[ct][r] / rsum[r];
            const int srow = blk * WW + qloc + r + 8 * hi;
            const int col = h * DH + ct * 16 + ln;
            O[((size_t)b * SS + srow) * DD + col] = f2bf(val);
        }
}

// ---------------------------------------------------------------------------
// Kernel 5: out-proj + residual, B tiles staged via Tensor Data Mover.
// A: attn bf16 [16384,1024]. Bw: WoT bf16 [1024,1024] (N,K). out f32.
// ---------------------------------------------------------------------------
__global__ void __launch_bounds__(128)
outproj_kernel(const unsigned short* __restrict__ A,
               const unsigned short* __restrict__ Bw,
               const float* __restrict__ x, float* __restrict__ out) {
    const int K = DD, N = DD, NIT = DD / 32;
    __shared__ unsigned short Bt[2][64 * 32];

    const int w = threadIdx.x >> 5, lane = threadIdx.x & 31;
    const int ln = lane & 15, hi = lane >> 4;
    const int rowA = blockIdx.y * 64 + w * 16 + ln;
    const int colBase = blockIdx.x * 64;
    const unsigned short* arow = A + (size_t)rowA * K;

    // TDM: DMA a 64(N-rows) x 32(K) bf16 tile into LDS; 2D descriptor.
    auto tdm_stage = [&](int buf, int ks) {
        const unsigned long long ga =
            (unsigned long long)(Bw + (size_t)colBase * K + ks);
        const unsigned int ldsa = lds_off(&Bt[buf][0]);
        u32x4 g0;
        g0.x = 0x1u;                                     // count=1 (valid D#)
        g0.y = ldsa;                                     // lds_addr
        g0.z = (unsigned int)(ga & 0xFFFFFFFFu);         // global_addr[31:0]
        g0.w = (unsigned int)((ga >> 32) & 0x1FFFFFFu)   // global_addr[56:32]
               | (2u << 30);                             // type=2 ("image")
        const unsigned int TD0 = 1u << 20, TD1 = 1u << 20;  // ample tensor dims
        i32x8 g1;
        g1[0] = 0x10000;                                 // data_size=1 (2B)
        g1[1] = (int)((TD0 & 0xFFFFu) << 16);            // tensor_dim0[15:0]
        g1[2] = (int)((TD0 >> 16) | ((TD1 & 0xFFFFu) << 16));
        g1[3] = (int)((TD1 >> 16) | (32u << 16));        // tile_dim0=32
        g1[4] = 64;                                      // tile_dim1=64, dim2=0
        g1[5] = (int)(unsigned)K;                        // dim0_stride lo32
        g1[6] = 0;
        g1[7] = 0;
        i32x4 z4 = (i32x4){};
        i32x8 z8 = (i32x8){};
        __builtin_amdgcn_tensor_load_to_lds(g0, g1, z4, z4, z8, 0);
    };

    v8f acc[4];
    for (int t = 0; t < 4; ++t) acc[t] = (v8f){};

    if (threadIdx.x < 32) tdm_stage(0, 0);               // wave 0 drives the TDM
#pragma unroll 1
    for (int i = 0; i < NIT; ++i) {
        const int cur = i & 1;
        if (threadIdx.x < 32) {
            if (i + 1 < NIT) {
                tdm_stage(cur ^ 1, (i + 1) * 32);
                __builtin_amdgcn_s_wait_tensorcnt(1);    // previous tile done
            } else {
                __builtin_amdgcn_s_wait_tensorcnt(0);
            }
        }
        __syncthreads();
        const int ks = i * 32;
        v16bf af = load_frag(arow + ks + 16 * hi, arow + ks + 8 + 16 * hi);
#pragma unroll
        for (int t = 0; t < 4; ++t) {
            const unsigned short* bp = &Bt[cur][(t * 16 + ln) * 32 + 16 * hi];
            v16bf bf = load_frag(bp, bp + 8);
            acc[t] = wmma_bf16(af, bf, acc[t]);
        }
        __syncthreads();
    }
    const int crow0 = blockIdx.y * 64 + w * 16;
#pragma unroll
    for (int t = 0; t < 4; ++t)
#pragma unroll
        for (int r = 0; r < 8; ++r) {
            const size_t m = crow0 + r + 8 * hi;
            const size_t n = colBase + t * 16 + ln;
            out[m * N + n] = x[m * N + n] + acc[t][r];
        }
}

// ---------------------------------------------------------------------------
extern "C" void kernel_launch(void* const* d_in, const int* in_sizes, int n_in,
                              void* d_out, int out_size, void* d_ws, size_t ws_size,
                              hipStream_t stream) {
    (void)in_sizes; (void)n_in; (void)out_size; (void)ws_size;
    const float* x    = (const float*)d_in[0];
    const float* g    = (const float*)d_in[1];
    const float* Wqkv = (const float*)d_in[2];
    const float* Wo   = (const float*)d_in[3];
    float* out = (float*)d_out;

    char* ws = (char*)d_ws;
    const size_t xnBytes    = (size_t)BB * SS * DD * 2;   // 32 MB
    const size_t qkvBytes   = (size_t)BB * SS * D3 * 2;   // 96 MB
    const size_t attnBytes  = (size_t)BB * SS * DD * 2;   // 32 MB
    const size_t wqkvTBytes = (size_t)DD * D3 * 2;        //  6 MB
    unsigned short* xn    = (unsigned short*)(ws);
    unsigned short* qkv   = (unsigned short*)(ws + xnBytes);
    unsigned short* attnO = (unsigned short*)(ws + xnBytes + qkvBytes);
    unsigned short* wqkvT = (unsigned short*)(ws + xnBytes + qkvBytes + attnBytes);
    unsigned short* woT   = (unsigned short*)(ws + xnBytes + qkvBytes + attnBytes + wqkvTBytes);

    rmsnorm_bf16_kernel<<<BB * SS, 256, 0, stream>>>(x, g, xn);
    transpose_bf16_kernel<<<(DD * D3) / 256, 256, 0, stream>>>(Wqkv, wqkvT, DD, D3);
    transpose_bf16_kernel<<<(DD * DD) / 256, 256, 0, stream>>>(Wo, woT, DD, DD);
    qkv_gemm_kernel<<<dim3(D3 / 64, (BB * SS) / 64), 128, 0, stream>>>(xn, wqkvT, qkv);
    attn_kernel<<<BB * HH * NB * 2, 128, 0, stream>>>(qkv, attnO);
    outproj_kernel<<<dim3(DD / 64, (BB * SS) / 64), 128, 0, stream>>>(attnO, woT, x, out);
}
